// Indexer_15333033247350
// MI455X (gfx1250) — matrix operand order repaired and checked by reading, not compiled, not run on hardware
//
#include <hip/hip_runtime.h>
#include <hip/hip_bf16.h>

#define S_    2048
#define DIM_  7168
#define QR_   1536
#define H_    64
#define D_    128
#define RD_   64
#define TOPK_ 2048
#define QKC   64          // K-chunk for LDS-staged GEMMs

#define USE_ASYNC_LDS 1

typedef __attribute__((ext_vector_type(16))) _Float16 v16h;
typedef __attribute__((ext_vector_type(8)))  float    v8f;

union Frag16 { v16h v; float4 f4[2]; };

// A fragment: 16(M) x 32(K) tile of row-major MxK. lane L: M=L%16,
// halves 0..7 = K[kb..kb+7], halves 8..15 = K[kb+16..kb+23], kb=8*(L/16)
__device__ __forceinline__ v16h load_frag_a(const _Float16* __restrict__ p0, int ld, int lane) {
  const _Float16* p = p0 + (lane & 15) * ld + ((lane >> 4) << 3);
  Frag16 f;
  f.f4[0] = *reinterpret_cast<const float4*>(p);
  f.f4[1] = *reinterpret_cast<const float4*>(p + 16);
  return f.v;
}

// B fragment: 32(K) x 16(N), supplied as B^T row-major (N x K).
// lane L: N=L%16 ; halves 0..7 = K[kb..kb+7], halves 8..15 = K[kb+16..kb+23], kb=8*(L/16)
__device__ __forceinline__ v16h load_frag_b(const _Float16* __restrict__ p0, int ld, int lane) {
  const _Float16* p = p0 + (lane & 15) * ld + ((lane >> 4) << 3);
  Frag16 f;
  f.f4[0] = *reinterpret_cast<const float4*>(p);
  f.f4[1] = *reinterpret_cast<const float4*>(p + 16);
  return f.v;
}

__device__ __forceinline__ v8f wmma_f16(v16h a, v16h b, v8f c) {
  return __builtin_amdgcn_wmma_f32_16x16x32_f16(false, a, false, b, (short)0, c, false, false);
}

// ---- async global->LDS staging (VGLOBAL async encoding; ASYNCcnt tracked) ----
__device__ __forceinline__ void async_ld16(void* lds, const void* g) {
#if USE_ASYNC_LDS
  unsigned loff = (unsigned)(unsigned long long)(uintptr_t)lds;  // generic->LDS offset (addr[31:0])
  asm volatile("global_load_async_to_lds_b128 %0, %1, off" :: "v"(loff), "v"(g) : "memory");
#else
  *(float4*)lds = *(const float4*)g;
#endif
}
__device__ __forceinline__ void async_wait0() {
#if USE_ASYNC_LDS
  asm volatile("s_wait_asynccnt 0x0" ::: "memory");
#endif
}

// ---------------- prep kernels ----------------
__global__ void f32_to_f16_kernel(const float* __restrict__ in, _Float16* __restrict__ out, int n) {
  int i = blockIdx.x * blockDim.x + threadIdx.x;
  if (i < n) out[i] = (_Float16)in[i];
}

// LDS-tiled transpose+convert: in (K x N f32, row-major) -> out (N x K f16, row-major).
__global__ __launch_bounds__(256) void transpose_f32_to_f16_tiled(const float* __restrict__ in,
                                                                  _Float16* __restrict__ out,
                                                                  int K, int N) {
  __shared__ float t[64][65];
  const int k0 = blockIdx.x * 64, n0 = blockIdx.y * 64;
  for (int u = threadIdx.x; u < 64 * 64; u += 256) {
    int r = u >> 6, cc = u & 63;                       // r: k-offset, cc: n-offset
    t[r][cc] = in[(size_t)(k0 + r) * N + n0 + cc];
  }
  __syncthreads();
  for (int u = threadIdx.x; u < 64 * 64; u += 256) {
    int r = u >> 6, cc = u & 63;                       // r: n-offset, cc: k-offset
    out[(size_t)(n0 + r) * K + k0 + cc] = (_Float16)t[cc][r];
  }
}

// ---------------- per-wave epilogue helpers on a 16x128 f32 LDS strip ----------------
__device__ __forceinline__ void rope_tile(float* tile, const float* __restrict__ cosb,
                                          const float* __restrict__ sinb, int s0, int lane) {
  float nv[32];
  int c = 0;
  for (int u = lane; u < 16 * RD_; u += 32) {
    int r = u >> 6, d = u & 63;
    const float* row = tile + r * D_;
    float v  = row[d];
    float vr = (d < 32) ? -row[d + 32] : row[d - 32];
    int s = s0 + r;
    nv[c++] = v * cosb[s * RD_ + d] + vr * sinb[s * RD_ + d];
  }
  __syncthreads();
  c = 0;
  for (int u = lane; u < 16 * RD_; u += 32) {
    int r = u >> 6, d = u & 63;
    tile[r * D_ + d] = nv[c++];
  }
}

__device__ __forceinline__ void hadamard_tile(float* tile, int lane) {
  for (int h2 = 1; h2 < D_; h2 <<= 1) {
    __syncthreads();
    for (int u = lane; u < 16 * (D_ / 2); u += 32) {
      int r = u >> 6, p = u & 63;
      int i = ((p & ~(h2 - 1)) << 1) | (p & (h2 - 1));
      float* row = tile + r * D_;
      float a = row[i], b = row[i + h2];
      row[i] = a + b;
      row[i + h2] = a - b;
    }
  }
  __syncthreads();
}

// ---------------- q path: 8 waves, 128x128 macro-tile per head, async LDS staging ----------------
__global__ __launch_bounds__(256)
__attribute__((amdgpu_waves_per_eu(2)))
void q_proj_kernel(const _Float16* __restrict__ qrh,   // S x QR
                   const _Float16* __restrict__ wqbT,  // (H*D) x QR
                   const float* __restrict__ cosb,
                   const float* __restrict__ sinb,
                   _Float16* __restrict__ qq) {        // H x S x D
  __shared__ alignas(16) _Float16 As[2][128 * QKC];
  __shared__ alignas(16) _Float16 Bs[2][128 * QKC];
  __shared__ alignas(16) float epi[128 * D_];
  const int tid  = threadIdx.x;
  const int lane = tid & 31;
  const int w    = tid >> 5;          // 8 waves
  const int i0   = blockIdx.x * 128;  // sequence rows
  const int h    = blockIdx.y;

  const _Float16* Ag = qrh + (size_t)i0 * QR_;
  const _Float16* Bg = wqbT + (size_t)(h * D_) * QR_;

  auto stageA = [&](int kc, int buf) {
    for (int c = tid; c < 128 * (QKC / 8); c += 256) {  // 16B chunks: 8/row, 4/thread
      int r = c >> 3, q8 = c & 7;
      async_ld16(&As[buf][r * QKC + q8 * 8], Ag + (size_t)r * QR_ + kc + q8 * 8);
    }
  };
  auto stageB = [&](int kc, int buf) {
    for (int c = tid; c < 128 * (QKC / 8); c += 256) {
      int r = c >> 3, q8 = c & 7;
      async_ld16(&Bs[buf][r * QKC + q8 * 8], Bg + (size_t)r * QR_ + kc + q8 * 8);
    }
  };

  v8f zero = {};
  v8f c[8];
#pragma unroll
  for (int t = 0; t < 8; ++t) c[t] = zero;

  stageA(0, 0);
  stageB(0, 0);
  const int NCH = QR_ / QKC;  // 24
  for (int ch = 0; ch < NCH; ++ch) {
    int buf = ch & 1;
    async_wait0();
    __syncthreads();                       // staging of `buf` complete for all threads
    if (ch + 1 < NCH) { stageA((ch + 1) * QKC, buf ^ 1); stageB((ch + 1) * QKC, buf ^ 1); }
    const _Float16* Abase = &As[buf][(w * 16) * QKC];

    // unroll-jam the whole 64-wide K-chunk: load 2 A frags + 16 B frags, then 16 WMMAs
    v16h a0 = load_frag_a(Abase, QKC, lane);
    v16h a1 = load_frag_a(Abase + 32, QKC, lane);
    v16h b0[8], b1[8];
#pragma unroll
    for (int t = 0; t < 8; ++t) {
      b0[t] = load_frag_b(&Bs[buf][(t * 16) * QKC], QKC, lane);
      b1[t] = load_frag_b(&Bs[buf][(t * 16) * QKC + 32], QKC, lane);
    }
#pragma unroll
    for (int t = 0; t < 8; ++t) c[t] = wmma_f16(a0, b0[t], c[t]);
#pragma unroll
    for (int t = 0; t < 8; ++t) c[t] = wmma_f16(a1, b1[t], c[t]);

    __syncthreads();                       // everyone done reading `buf` before it is restaged
  }

  float* tile = epi + (w * 16) * D_;
  const int mrow = (lane >> 4) << 3;
  const int ncol = lane & 15;
#pragma unroll
  for (int t = 0; t < 8; ++t)
#pragma unroll
    for (int r = 0; r < 8; ++r)
      tile[(mrow + r) * D_ + t * 16 + ncol] = c[t][r];
  __syncthreads();

  rope_tile(tile, cosb, sinb, i0 + w * 16, lane);
  hadamard_tile(tile, lane);

  const float hsc = 0.08838834764831845f;  // 128^-0.5
  for (int u = lane; u < 16 * D_; u += 32) {
    int r = u >> 7, d = u & 127;
    qq[((size_t)h * S_ + (i0 + w * 16 + r)) * D_ + d] = (_Float16)(tile[r * D_ + d] * hsc);
  }
}

// ---------------- k & w path: one wave / 16-row strip, register-double-buffered ----------------
__global__ __launch_bounds__(32)
__attribute__((amdgpu_waves_per_eu(1)))
void kw_kernel(const _Float16* __restrict__ xh,   // S x DIM
               const _Float16* __restrict__ wkT,  // D x DIM
               const _Float16* __restrict__ wpT,  // H x DIM
               const float* __restrict__ cosb,
               const float* __restrict__ sinb,
               const float* __restrict__ knw,
               const float* __restrict__ knb,
               _Float16* __restrict__ kk,   // S x D
               float* __restrict__ wout) {  // S x H
  __shared__ alignas(16) float tile[16 * D_];
  __shared__ float mu_s[16], rs_s[16];
  const int lane = threadIdx.x;
  const int s0 = blockIdx.x * 16;

  v8f zero = {};
  v8f ck[8], cw[4];
#pragma unroll
  for (int t = 0; t < 8; ++t) ck[t] = zero;
#pragma unroll
  for (int t = 0; t < 4; ++t) cw[t] = zero;

  const _Float16* Abase = xh + (size_t)s0 * DIM_;

  v16h a[2];
  v16h bk[2][8];
  v16h bw[2][4];
  auto loadStage = [&](int st, int ks) {
    __builtin_prefetch(Abase + ks + 64, 0, 3);
    a[st] = load_frag_a(Abase + ks, DIM_, lane);
#pragma unroll
    for (int t = 0; t < 8; ++t) bk[st][t] = load_frag_b(wkT + (size_t)(t * 16) * DIM_ + ks, DIM_, lane);
#pragma unroll
    for (int t = 0; t < 4; ++t) bw[st][t] = load_frag_b(wpT + (size_t)(t * 16) * DIM_ + ks, DIM_, lane);
  };
  auto compStage = [&](int st) {
#pragma unroll
    for (int t = 0; t < 8; ++t) ck[t] = wmma_f16(a[st], bk[st][t], ck[t]);
#pragma unroll
    for (int t = 0; t < 4; ++t) cw[t] = wmma_f16(a[st], bw[st][t], cw[t]);
  };

  loadStage(0, 0);
  for (int ks = 0; ks < DIM_; ks += 64) {        // DIM_ % 64 == 0
    if (ks + 32 < DIM_) loadStage(1, ks + 32);   // overlap with compute of stage 0
    compStage(0);
    if (ks + 64 < DIM_) loadStage(0, ks + 64);   // overlap with compute of stage 1
    compStage(1);
  }

  const int mrow = (lane >> 4) << 3;
  const int ncol = lane & 15;

#pragma unroll
  for (int t = 0; t < 4; ++t)
#pragma unroll
    for (int r = 0; r < 8; ++r)
      wout[(size_t)(s0 + mrow + r) * H_ + t * 16 + ncol] = cw[t][r] * 0.125f;  // H^-0.5

#pragma unroll
  for (int t = 0; t < 8; ++t)
#pragma unroll
    for (int r = 0; r < 8; ++r)
      tile[(mrow + r) * D_ + t * 16 + ncol] = ck[t][r];
  __syncthreads();

  if (lane < 16) {
    const float* row = tile + lane * D_;
    float m = 0.f;
    for (int d = 0; d < D_; ++d) m += row[d];
    m *= (1.0f / D_);
    float v = 0.f;
    for (int d = 0; d < D_; ++d) { float t0 = row[d] - m; v += t0 * t0; }
    v *= (1.0f / D_);
    mu_s[lane] = m;
    rs_s[lane] = rsqrtf(v + 1e-6f);
  }
  __syncthreads();
  for (int u = lane; u < 16 * D_; u += 32) {
    int r = u >> 7, d = u & 127;
    tile[u] = (tile[u] - mu_s[r]) * rs_s[r] * knw[d] + knb[d];
  }
  __syncthreads();

  rope_tile(tile, cosb, sinb, s0, lane);
  hadamard_tile(tile, lane);

  const float hsc = 0.08838834764831845f;
  for (int u = lane; u < 16 * D_; u += 32) {
    int r = u >> 7, d = u & 127;
    kk[(size_t)(s0 + r) * D_ + d] = (_Float16)(tile[u] * hsc);
  }
}

// ---------------- scores: 16 waves, 128x64 causal macro-tile, async-staged qq heads ----------------
__global__ __launch_bounds__(512)
__attribute__((amdgpu_waves_per_eu(2)))
void scores_kernel(const _Float16* __restrict__ qq,  // H x S x D
                   const _Float16* __restrict__ kk,  // S x D
                   const float* __restrict__ w,      // S x H
                   float* __restrict__ scores) {     // S x S
  const int bi = blockIdx.y;          // 128 rows each
  const int bj = blockIdx.x;          // 64 cols each
  const int i0 = bi * 128, j0 = bj * 64;
  if (j0 > i0 + 127) return;          // fully masked macro-tile

  __shared__ alignas(16) _Float16 As[2][128 * D_];  // 2 x 32KB
  __shared__ float wl[128 * H_];                    // 32KB
  const int tid  = threadIdx.x;
  const int lane = tid & 31;
  const int wv   = tid >> 5;          // 16 waves
  const int wi   = wv & 7;            // row strip  (8 x 16 = 128)
  const int wjg  = wv >> 3;           // col group  (2 x 32 = 64)

  // loop-invariant kk B-fragments held in registers (two 16-col sub-tiles, 4 K-steps)
  v16h breg[2][4];
#pragma unroll
  for (int jt = 0; jt < 2; ++jt)
#pragma unroll
    for (int ks = 0; ks < 4; ++ks)
      breg[jt][ks] = load_frag_b(kk + (size_t)(j0 + wjg * 32 + jt * 16) * D_ + ks * 32, D_, lane);

  for (int u = tid; u < 128 * H_; u += 512)
    wl[u] = w[(size_t)(i0 + (u >> 6)) * H_ + (u & 63)];

  auto stageA = [&](int h, int buf) {
    const _Float16* Ag = qq + ((size_t)h * S_ + i0) * D_;
    for (int c = tid; c < 128 * 16; c += 512) {  // 16B chunks: 16/row, 4/thread
      int r = c >> 4, q8 = c & 15;
      async_ld16(&As[buf][r * D_ + q8 * 8], Ag + (size_t)r * D_ + q8 * 8);
    }
  };

  stageA(0, 0);

  float acc0[8] = {}, acc1[8] = {};
  const float scale = 0.08838834764831845f;  // D^-0.5
  const int mrow = (lane >> 4) << 3;
  const int ncol = lane & 15;

  for (int h = 0; h < H_; ++h) {
    int buf = h & 1;
    async_wait0();
    __syncthreads();                   // head tile staged for all threads (also covers wl on h==0)
    if (h + 1 < H_) stageA(h + 1, buf ^ 1);

    // batch all 4 A-fragment LDS loads, then 8 WMMAs (single dscnt wait)
    const _Float16* Abase = &As[buf][(wi * 16) * D_];
    v16h av[4];
#pragma unroll
    for (int ks = 0; ks < 4; ++ks) av[ks] = load_frag_a(Abase + ks * 32, D_, lane);

    v8f c0 = {}, c1 = {};
#pragma unroll
    for (int ks = 0; ks < 4; ++ks) {
      c0 = wmma_f16(av[ks], breg[0][ks], c0);
      c1 = wmma_f16(av[ks], breg[1][ks], c1);
    }
#pragma unroll
    for (int r = 0; r < 8; ++r) {
      float wgt = wl[(wi * 16 + mrow + r) * H_ + h];
      acc0[r] += wgt * fmaxf(c0[r] * scale, 0.0f);
      acc1[r] += wgt * fmaxf(c1[r] * scale, 0.0f);
    }
    __syncthreads();                   // all waves done reading `buf` before it is restaged
  }

#pragma unroll
  for (int r = 0; r < 8; ++r) {
    int gi = i0 + wi * 16 + mrow + r;
    scores[(size_t)gi * S_ + j0 + wjg * 32 + ncol]      = acc0[r];
    scores[(size_t)gi * S_ + j0 + wjg * 32 + 16 + ncol] = acc1[r];
  }
}

// ---------------- top-k (TOPK==S): full descending bitonic sort per causal row ----------------
__global__ __launch_bounds__(256) void topk_kernel(const float* __restrict__ scores,
                                                   float* __restrict__ vals_out,
                                                   int* __restrict__ idx_out) {
  __shared__ float sv[S_];
  __shared__ int   si[S_];
  const int row = blockIdx.x, tid = threadIdx.x;

  for (int j = tid; j < S_; j += 256) {
    float v = (j <= row) ? scores[(size_t)row * S_ + j] : -INFINITY;
    sv[j] = v;
    si[j] = j;
  }
  __syncthreads();

  for (int k = 2; k <= S_; k <<= 1) {
    for (int jj = k >> 1; jj > 0; jj >>= 1) {
      for (int t = tid; t < S_ / 2; t += 256) {
        int i = ((t & ~(jj - 1)) << 1) | (t & (jj - 1));
        int ixj = i | jj;
        float a = sv[i], b = sv[ixj];
        bool descSeg = ((i & k) == 0);
        bool doSwap = descSeg ? (a < b) : (a > b);
        if (doSwap) {
          sv[i] = b; sv[ixj] = a;
          int ti = si[i]; si[i] = si[ixj]; si[ixj] = ti;
        }
      }
      __syncthreads();
    }
  }

  for (int j = tid; j < S_; j += 256) {
    float v = sv[j];
    bool bad = (v == -INFINITY);
    vals_out[(size_t)row * TOPK_ + j] = bad ? 0.0f : v;
    idx_out[(size_t)row * TOPK_ + j]  = bad ? -1 : si[j];
  }
}

// ---------------- host side ----------------
extern "C" void kernel_launch(void* const* d_in, const int* in_sizes, int n_in,
                              void* d_out, int out_size, void* d_ws, size_t ws_size,
                              hipStream_t stream) {
  const float* x    = (const float*)d_in[0];
  const float* qr   = (const float*)d_in[1];
  const float* cosb = (const float*)d_in[2];
  const float* sinb = (const float*)d_in[3];
  const float* wq_b = (const float*)d_in[4];
  const float* wk   = (const float*)d_in[5];
  const float* knw  = (const float*)d_in[6];
  const float* knb  = (const float*)d_in[7];
  const float* wp   = (const float*)d_in[8];

  size_t off = 0;
  auto carve = [&](size_t bytes) -> void* {
    void* p = (void*)((char*)d_ws + off);
    off += (bytes + 255) & ~(size_t)255;
    return p;
  };
  _Float16* qrh  = (_Float16*)carve((size_t)S_ * QR_ * 2);
  _Float16* xh   = (_Float16*)carve((size_t)S_ * DIM_ * 2);
  _Float16* wqbT = (_Float16*)carve((size_t)H_ * D_ * QR_ * 2);
  _Float16* wkT  = (_Float16*)carve((size_t)D_ * DIM_ * 2);
  _Float16* wpT  = (_Float16*)carve((size_t)H_ * DIM_ * 2);
  _Float16* qq   = (_Float16*)carve((size_t)H_ * S_ * D_ * 2);
  _Float16* kkh  = (_Float16*)carve((size_t)S_ * D_ * 2);
  float*    wbuf = (float*)carve((size_t)S_ * H_ * 4);
  float*    sc   = (float*)carve((size_t)S_ * S_ * 4);

  f32_to_f16_kernel<<<(S_ * QR_ + 255) / 256, 256, 0, stream>>>(qr, qrh, S_ * QR_);
  f32_to_f16_kernel<<<(S_ * DIM_ + 255) / 256, 256, 0, stream>>>(x, xh, S_ * DIM_);
  transpose_f32_to_f16_tiled<<<dim3(QR_ / 64, (H_ * D_) / 64), 256, 0, stream>>>(wq_b, wqbT, QR_, H_ * D_);
  transpose_f32_to_f16_tiled<<<dim3(DIM_ / 64, D_ / 64), 256, 0, stream>>>(wk, wkT, DIM_, D_);
  transpose_f32_to_f16_tiled<<<dim3(DIM_ / 64, H_ / 64), 256, 0, stream>>>(wp, wpT, DIM_, H_);

  q_proj_kernel<<<dim3(S_ / 128, H_), 256, 0, stream>>>(qrh, wqbT, cosb, sinb, qq);
  kw_kernel<<<S_ / 16, 32, 0, stream>>>(xh, wkT, wpT, cosb, sinb, knw, knb, kkh, wbuf);

  scores_kernel<<<dim3(S_ / 64, S_ / 128), 512, 0, stream>>>(qq, kkh, wbuf, sc);

  float* vals_out = (float*)d_out;
  int*   idx_out  = (int*)((float*)d_out + (size_t)S_ * TOPK_);
  topk_kernel<<<S_, 256, 0, stream>>>(sc, vals_out, idx_out);
}
